// NeuralGasEngine_37752762532592
// MI455X (gfx1250) — matrix operand order repaired and checked by reading, not compiled, not run on hardware
//
#include <hip/hip_runtime.h>
#include <math.h>

#define N_CELLS 8192
#define HID     128
#define IN_DIM  64
#define OUT_DIMV 64
#define NF      8
#define FS      1024
#define TOPK    8

// ws float-offset layout
#define O_SIG   0        // 128  signal
#define O_FSUM  128      // 1024 per-faction sums of prototypes (8 x 128)
#define O_CORR  1152     // 1024 per-faction sum of P over nmask rows (excl. bmu1/bmu2)
#define O_CNT   2176     // 8    per-faction nmask counts (excl. bmu1/bmu2)
#define O_DIST  2184     // 8192 distances
#define O_TOPI  10376    // 8    top indices (stored as int)
#define O_TOPD  10384    // 8    top distances
#define O_NMASK 10392    // 8192 nmask
#define ZERO_BEG 128
#define ZERO_CNT 2056

typedef __attribute__((ext_vector_type(16))) _Float16 v16h;
typedef __attribute__((ext_vector_type(8)))  float    v8f;

// ---------- K0: zero accumulators (re-run every call; graph-replay safe) ----------
__global__ void k_zero(float* ws) {
  int i = blockIdx.x * blockDim.x + threadIdx.x;
  if (i < ZERO_CNT) ws[ZERO_BEG + i] = 0.f;
}

// ---------- K1: signal = x @ in_w.T + in_b  (128 outputs) ----------
__global__ void k_signal(const float* __restrict__ x, const float* __restrict__ in_w,
                         const float* __restrict__ in_b, float* __restrict__ ws) {
  int j = threadIdx.x;
  float acc = in_b[j];
#pragma unroll 8
  for (int d = 0; d < IN_DIM; ++d) acc += in_w[j * IN_DIM + d] * x[d];
  ws[O_SIG + j] = acc;
}

// ---------- K2: fused dists + per-faction prototype sums (the 4 MB pass) ----------
// block = 256 (8 waves), 64 contiguous rows per block (one faction per block)
__global__ void k_dist(const float* __restrict__ P, float* __restrict__ ws) {
  const int tid  = threadIdx.x;
  const int lane = tid & 31;
  const int wave = tid >> 5;
  const int base = blockIdx.x * 64;
  const int f    = base >> 10;
  const float s0 = ws[O_SIG + lane];
  const float s1 = ws[O_SIG + lane + 32];
  const float s2 = ws[O_SIG + lane + 64];
  const float s3 = ws[O_SIG + lane + 96];
  float fa0 = 0.f, fa1 = 0.f, fa2 = 0.f, fa3 = 0.f;
  for (int r = 0; r < 8; ++r) {
    int i = base + wave * 8 + r;
    const float* Pr = P + (size_t)i * HID;
    __builtin_prefetch(Pr + 8 * HID, 0, 0);   // global_prefetch_b8
    float v0 = Pr[lane], v1 = Pr[lane + 32], v2 = Pr[lane + 64], v3 = Pr[lane + 96];
    fa0 += v0; fa1 += v1; fa2 += v2; fa3 += v3;
    float d0 = v0 - s0, d1 = v1 - s1, d2 = v2 - s2, d3 = v3 - s3;
    float dsq = d0 * d0 + d1 * d1 + d2 * d2 + d3 * d3;
#pragma unroll
    for (int off = 16; off > 0; off >>= 1) dsq += __shfl_xor(dsq, off, 32);
    if (lane == 0) ws[O_DIST + i] = dsq;
  }
  atomicAdd(&ws[O_FSUM + f * HID + lane     ], fa0);
  atomicAdd(&ws[O_FSUM + f * HID + lane + 32], fa1);
  atomicAdd(&ws[O_FSUM + f * HID + lane + 64], fa2);
  atomicAdd(&ws[O_FSUM + f * HID + lane + 96], fa3);
}

// ---------- K3: top-8 smallest distances (stable tie-break = stable argsort) ----------
__global__ void k_topk(float* ws) {
  __shared__ float sv[256];
  __shared__ int   si[256];
  __shared__ int   chosen[TOPK];
  const int tid = threadIdx.x;
  int* topi = (int*)&ws[O_TOPI];
  for (int p = 0; p < TOPK; ++p) {
    float bv = 3.4e38f; int bi = 0x7fffffff;
    for (int i = tid; i < N_CELLS; i += 256) {
      bool skip = false;
      for (int q = 0; q < p; ++q) if (chosen[q] == i) skip = true;
      if (skip) continue;
      float v = ws[O_DIST + i];
      if (v < bv || (v == bv && i < bi)) { bv = v; bi = i; }
    }
    sv[tid] = bv; si[tid] = bi;
    __syncthreads();
    for (int off = 128; off > 0; off >>= 1) {
      if (tid < off) {
        float ov = sv[tid + off]; int oi = si[tid + off];
        if (ov < sv[tid] || (ov == sv[tid] && oi < si[tid])) { sv[tid] = ov; si[tid] = oi; }
      }
      __syncthreads();
    }
    if (tid == 0) { chosen[p] = si[0]; topi[p] = si[0]; ws[O_TOPD + p] = sv[0]; }
    __syncthreads();
  }
}

// ---------- K4: nmask row (only row bmu1 of edges/ages!) + faction corrections ----------
__global__ void k_nmask(const float* __restrict__ P, const float* __restrict__ edges,
                        const float* __restrict__ ages, float* __restrict__ ws) {
  int i = blockIdx.x * blockDim.x + threadIdx.x;
  const int* topi = (const int*)&ws[O_TOPI];
  int b1 = topi[0], b2 = topi[1];
  float eij = edges[(size_t)b1 * N_CELLS + i];
  float aij = ages [(size_t)b1 * N_CELLS + i];
  if (i == b2) { eij = 1.f; aij = 0.f; }               // set edge, reset age
  float age = aij + ((i == b1) ? 2.f : 1.f);           // row add +1; diag gets column add too
  float ef  = (age > 50.f) ? 0.f : eij;                // prune old
  float nm  = (ef > 0.f) ? 1.f : 0.f;
  ws[O_NMASK + i] = nm;
  if (nm > 0.f && i != b1 && i != b2) {                // bmu1/bmu2 handled analytically
    int f = i >> 10;
    atomicAdd(&ws[O_CNT + f], 1.f);
    const float* Pr = P + (size_t)i * HID;
    for (int d = 0; d < HID; ++d) atomicAdd(&ws[O_CORR + f * HID + d], Pr[d]);
  }
}

// ---------- WMMA GEMV layer: hout[0..outDim) = (relu)(W(outDim x128) . hin + bias) ----------
// Single wave32. A = weight tile (16x32 f16, ISA lane layout), B = broadcast activation chunk.
__device__ __forceinline__ void wmma_mlp(const float* __restrict__ W,
                                         const float* __restrict__ bias,
                                         const float* __restrict__ hin,
                                         float* __restrict__ hout,
                                         int outDim, bool doRelu) {
  const int lane = threadIdx.x;
  const int M    = lane & 15;
  const int hi   = (lane >= 16) ? 1 : 0;
  const int nj   = outDim >> 4;
  for (int jt = 0; jt < nj; ++jt) {
    v8f c = {};
    for (int ks = 0; ks < 4; ++ks) {      // K = 128 in steps of 32
      v16h a, b;
#pragma unroll
      for (int e = 0; e < 16; ++e) {
        // 16-bit A 16x32 layout: lanes 0-15 hold K {0..7,16..23}; lanes 16-31 hold K {8..15,24..31}
        int K = hi ? ((e < 8) ? e + 8 : e + 16) : ((e < 8) ? e : e + 8);
        a[e] = (_Float16)W[(jt * 16 + M) * HID + ks * 32 + K];
        // B 32x16: same K order as A; broadcast across N columns
        b[e] = (_Float16)hin[ks * 32 + K];
      }
      c = __builtin_amdgcn_wmma_f32_16x16x32_f16(false, a, false, b, (short)0, c, false, false);
    }
    // f32 C/D: VGPR r -> M = r + 8*hi, N = lane&15; all columns identical -> read N==0 lanes
    if (lane == 0) {
#pragma unroll
      for (int r = 0; r < 8; ++r) {
        float v = c[r] + bias[jt * 16 + r];
        hout[jt * 16 + r] = doRelu ? fmaxf(v, 0.f) : v;
      }
    }
    if (lane == 16) {
#pragma unroll
      for (int r = 0; r < 8; ++r) {
        float v = c[r] + bias[jt * 16 + 8 + r];
        hout[jt * 16 + 8 + r] = doRelu ? fmaxf(v, 0.f) : v;
      }
    }
  }
  __syncthreads();
}

// ---------- K5: single-wave finale ----------
__global__ void __launch_bounds__(32) k_final(
    const float* __restrict__ P, const int* __restrict__ stepp,
    const float* __restrict__ ea_w1, const float* __restrict__ ea_b1,
    const float* __restrict__ ea_w2, const float* __restrict__ ea_b2,
    const float* __restrict__ eg_w1, const float* __restrict__ eg_b1,
    const float* __restrict__ eg_w2, const float* __restrict__ eg_b2,
    const float* __restrict__ out_w, const float* __restrict__ out_b,
    float* __restrict__ ws, float* __restrict__ out)
{
  __shared__ float sh[HID];       // winner_h
  __shared__ float h1s[HID];      // hidden activations
  __shared__ float aout[OUT_DIMV];
  __shared__ float gout[OUT_DIMV];
  __shared__ float comb[HID];
  const int lane = threadIdx.x;
  const int step = stepp[0];
  const float eps_w = fmaxf(0.05f, 0.3f * expf(-(float)step / 200.f));
  const float eps_n = 0.01f * eps_w;

  float s4[4];
#pragma unroll
  for (int j = 0; j < 4; ++j) s4[j] = ws[O_SIG + lane + 32 * j];

  const int* topi = (const int*)&ws[O_TOPI];
  int top[8]; float td[8];
#pragma unroll
  for (int k = 0; k < 8; ++k) { top[k] = topi[k]; td[k] = ws[O_TOPD + k]; }
  const int b1 = top[0], b2 = top[1];
  const float nm1 = ws[O_NMASK + b1];

  // pre-faction proto rows for the 8 needed indices + deltas for bmu1/bmu2
  float q[8][4], d1[4], d2[4];
  for (int k = 0; k < 8; ++k) {
    int idx = top[k];
    const float* Pr = P + (size_t)idx * HID;
    float nm = ws[O_NMASK + idx];
#pragma unroll
    for (int j = 0; j < 4; ++j) {
      float p = Pr[lane + 32 * j];
      float t;
      if (idx == b1) {
        t = p + eps_w * (s4[j] - p);
        if (nm1 > 0.f) t = t + eps_n * (s4[j] - t);
      } else if (idx == b2) {
        t = p + eps_n * (s4[j] - p);
        t = t + eps_n * (s4[j] - t);
      } else {
        t = p + nm * eps_n * (s4[j] - p);
      }
      q[k][j] = t;
      if (k == 0) d1[j] = t - p;
      if (k == 1) d2[j] = t - p;
    }
  }

  // analytic faction means: base sums + eps_n*(cnt*s - corr) + bmu deltas
  const int f1 = b1 >> 10, f2 = b2 >> 10;
  float fm[8][4];
  for (int f = 0; f < NF; ++f) {
    float c = ws[O_CNT + f];
#pragma unroll
    for (int j = 0; j < 4; ++j) {
      float v = ws[O_FSUM + f * HID + lane + 32 * j]
              + eps_n * (c * s4[j] - ws[O_CORR + f * HID + lane + 32 * j]);
      if (f == f1) v += d1[j];
      if (f == f2) v += d2[j];
      fm[f][j] = v * (1.f / (float)FS);
    }
  }
  float gop[4];
#pragma unroll
  for (int j = 0; j < 4; ++j) {
    float a = 0.f;
    for (int f = 0; f < NF; ++f) a += fm[f][j];
    gop[j] = a * 0.125f;
  }

  // faction sync + debate (only for the 8 rows we consume)
  const bool debate = step > 5;
  for (int k = 0; k < 8; ++k) {
    int idx = top[k]; int f = idx >> 10;
    bool dc = debate && ((idx & (FS - 1)) < (FS / 4));
#pragma unroll
    for (int j = 0; j < 4; ++j) {
      float bv = 0.85f * q[k][j] + 0.15f * fm[f][j];
      if (dc) bv = 0.85f * bv + 0.15f * gop[j];
      q[k][j] = bv;
    }
  }

  // winner_h -> LDS, then WMMA expert heads
#pragma unroll
  for (int j = 0; j < 4; ++j) sh[lane + 32 * j] = q[0][j];
  __syncthreads();
  wmma_mlp(ea_w1, ea_b1, sh,  h1s,  HID,      true );
  wmma_mlp(ea_w2, ea_b2, h1s, aout, OUT_DIMV, false);
  wmma_mlp(eg_w1, eg_b1, sh,  h1s,  HID,      true );
  wmma_mlp(eg_w2, eg_b2, h1s, gout, OUT_DIMV, false);

  // tension = mean((a-g)^2)
  float t2 = 0.f;
#pragma unroll
  for (int m = 0; m < 2; ++m) {
    float dg = aout[lane + 32 * m] - gout[lane + 32 * m];
    t2 += dg * dg;
  }
#pragma unroll
  for (int off = 16; off > 0; off >>= 1) t2 += __shfl_xor(t2, off, 32);

  // softmax(-dists[top]) weights; td[0] is the min distance
  float wk[8]; float wsum = 0.f;
#pragma unroll
  for (int k = 0; k < 8; ++k) { wk[k] = expf(td[0] - td[k]); wsum += wk[k]; }
  float inv = 1.f / wsum;
#pragma unroll
  for (int j = 0; j < 4; ++j) {
    float c = 0.f;
    for (int k = 0; k < 8; ++k) c += wk[k] * q[k][j];
    comb[lane + 32 * j] = c * inv;
  }
  __syncthreads();

  // output = combined @ out_w.T + out_b
#pragma unroll
  for (int m = 0; m < 2; ++m) {
    int jj = lane + 32 * m;
    float acc = out_b[jj];
    for (int d = 0; d < HID; ++d) acc += out_w[jj * HID + d] * comb[d];
    out[jj] = acc;
  }
  if (lane == 0) out[64] = t2 * (1.f / 64.f);
}

extern "C" void kernel_launch(void* const* d_in, const int* in_sizes, int n_in,
                              void* d_out, int out_size, void* d_ws, size_t ws_size,
                              hipStream_t stream) {
  const float* x     = (const float*)d_in[0];
  const int*   step  = (const int*)  d_in[1];
  const float* P     = (const float*)d_in[2];
  const float* edges = (const float*)d_in[3];
  const float* ages  = (const float*)d_in[4];
  const float* in_w  = (const float*)d_in[5];
  const float* in_b  = (const float*)d_in[6];
  const float* out_w = (const float*)d_in[7];
  const float* out_b = (const float*)d_in[8];
  const float* ea_w1 = (const float*)d_in[9];
  const float* ea_b1 = (const float*)d_in[10];
  const float* ea_w2 = (const float*)d_in[11];
  const float* ea_b2 = (const float*)d_in[12];
  const float* eg_w1 = (const float*)d_in[13];
  const float* eg_b1 = (const float*)d_in[14];
  const float* eg_w2 = (const float*)d_in[15];
  const float* eg_b2 = (const float*)d_in[16];
  float* ws  = (float*)d_ws;
  float* out = (float*)d_out;

  hipLaunchKernelGGL(k_zero,   dim3((ZERO_CNT + 255) / 256), dim3(256), 0, stream, ws);
  hipLaunchKernelGGL(k_signal, dim3(1),            dim3(HID), 0, stream, x, in_w, in_b, ws);
  hipLaunchKernelGGL(k_dist,   dim3(N_CELLS / 64), dim3(256), 0, stream, P, ws);
  hipLaunchKernelGGL(k_topk,   dim3(1),            dim3(256), 0, stream, ws);
  hipLaunchKernelGGL(k_nmask,  dim3(N_CELLS / 256),dim3(256), 0, stream, P, edges, ages, ws);
  hipLaunchKernelGGL(k_final,  dim3(1),            dim3(32),  0, stream, P, step,
                     ea_w1, ea_b1, ea_w2, ea_b2, eg_w1, eg_b1, eg_w2, eg_b2,
                     out_w, out_b, ws, out);
}
